// CrossAttention_49649821941786
// MI455X (gfx1250) — matrix operand order repaired and checked by reading, not compiled, not run on hardware
//
#include <hip/hip_runtime.h>

#define DIM   1024
#define HEADS 8
#define DH    128
#define NB    2
#define NQ    4096
#define NK    2048
#define ROWS  32
#define SSTR  (NK + 4)

typedef __attribute__((ext_vector_type(16))) __bf16 v16bf;
typedef __attribute__((ext_vector_type(2)))  __bf16 bf16x2;
typedef __attribute__((ext_vector_type(8)))  float  v8f;
typedef __attribute__((ext_vector_type(4)))  int    int4v;

typedef __attribute__((address_space(1))) int4v gint4;   // global int4
typedef __attribute__((address_space(3))) int4v lint4;   // LDS int4

// ---- gfx1250 async copy global->LDS (ASYNCcnt path) ------------------------
// Device-pass verified signature: (AS1 int4* src, AS3 int4* dst, imm, imm cpol)
#if defined(__HIP_DEVICE_COMPILE__)
# if __has_builtin(__builtin_amdgcn_global_load_async_to_lds_b128)
#  define ASYNC_CP_B128(gp, lp) \
    __builtin_amdgcn_global_load_async_to_lds_b128((gint4*)(gp), (lint4*)(lp), 0, 0)
# else
#  define ASYNC_CP_B128(gp, lp)                                                  \
    asm volatile("global_load_async_to_lds_b128 %0, %1, off"                     \
                 :: "v"((unsigned)(unsigned long long)(lp)),                     \
                    "v"((const void*)(gp))                                       \
                 : "memory")
# endif
#else
# define ASYNC_CP_B128(gp, lp) ((void)(gp), (void)(lp))   // host pass: no-op
#endif

__device__ inline void wait_async0() {
#if defined(__HIP_DEVICE_COMPILE__)
# if __has_builtin(__builtin_amdgcn_s_wait_asynccnt)
  __builtin_amdgcn_s_wait_asynccnt(0);
# else
  asm volatile("s_wait_asynccnt 0" ::: "memory");
# endif
#endif
}

// ---- WMMA fragment builders ------------------------------------------------
// A-matrix (16x32 bf16): VGPR v holds the K pair at
//   kb = (v&4 ? 16 : 0) + half*8 + (v&3)*2          (ISA 7.12.2, 16-bit A)
// B-matrix (32x16 bf16): lane holds column n = lane&15; VGPR v holds K pair at
//   kb = (v&4 ? 16 : 0) + half*8 + (v&3)*2          (same order as A)

__device__ inline v16bf frag_a_from_f32(const float* __restrict__ row, int kk, int half) {
  v16bf a;
#pragma unroll
  for (int v = 0; v < 8; ++v) {
    const int kb = kk + ((v & 4) ? 16 : 0) + half * 8 + (v & 3) * 2;
    const float2 f = *(const float2*)(row + kb);
    a[2 * v]     = (__bf16)f.x;
    a[2 * v + 1] = (__bf16)f.y;
  }
  return a;
}

__device__ inline v16bf frag_b_from_f32(const float* __restrict__ col, int kk, int half) {
  v16bf b;
#pragma unroll
  for (int v = 0; v < 8; ++v) {
    const int kb = kk + ((v & 4) ? 16 : 0) + half * 8 + (v & 3) * 2;
    const float2 f = *(const float2*)(col + kb);
    b[2 * v]     = (__bf16)f.x;
    b[2 * v + 1] = (__bf16)f.y;
  }
  return b;
}

__device__ inline v16bf frag_a_from_bf16(const __bf16* __restrict__ row, int kk, int half) {
  v16bf a;
#pragma unroll
  for (int v = 0; v < 8; ++v) {
    const int kb = kk + ((v & 4) ? 16 : 0) + half * 8 + (v & 3) * 2;
    const bf16x2 p = *(const bf16x2*)(row + kb);
    a[2 * v] = p[0];
    a[2 * v + 1] = p[1];
  }
  return a;
}

__device__ inline v16bf frag_b_from_bf16(const __bf16* __restrict__ col, int kk, int half) {
  v16bf b;
#pragma unroll
  for (int v = 0; v < 8; ++v) {
    const int kb = kk + ((v & 4) ? 16 : 0) + half * 8 + (v & 3) * 2;
    const bf16x2 p = *(const bf16x2*)(col + kb);
    b[2 * v] = p[0];
    b[2 * v + 1] = p[1];
  }
  return b;
}

__device__ inline v8f wmma_bf16(v16bf a, v16bf b, v8f c) {
  return __builtin_amdgcn_wmma_f32_16x16x32_bf16(false, a, false, b, (short)0, c,
                                                 false, false);
}

// ---- Projection: Y = X @ W^T + bias, bf16 output ---------------------------
// One wave per 16x16 output tile; 8 waves / 256 threads per block.
// VTRANS=true scatters the result as [b][h][d][n2] (for the P*V B operand).
template <bool VTRANS>
__global__ __launch_bounds__(256) void proj_kernel(
    const float* __restrict__ X, const float* __restrict__ W,
    const float* __restrict__ bias, __bf16* __restrict__ Y)
{
  const int tile = blockIdx.x * 8 + (threadIdx.x >> 5);
  const int lane = threadIdx.x & 31;
  const int half = lane >> 4;
  const int nl   = lane & 15;
  const int tn   = tile & 63;       // DIM/16 = 64 column tiles
  const int tm   = tile >> 6;
  const int row0 = tm * 16;
  const int col0 = tn * 16;

  const float* arow = X + (size_t)(row0 + nl) * DIM;
  const float* bcol = W + (size_t)(col0 + nl) * DIM;  // torch Linear: B = W row n

  v8f acc = {};
#pragma unroll 4
  for (int kk = 0; kk < DIM; kk += 32) {
    const v16bf a = frag_a_from_f32(arow, kk, half);
    const v16bf b = frag_b_from_f32(bcol, kk, half);
    acc = wmma_bf16(a, b, acc);
  }

  const float bi = bias[col0 + nl];
#pragma unroll
  for (int v = 0; v < 8; ++v) {
    const int m   = v + half * 8;      // D layout: M = v + 8*half, N = lane&15
    const int row = row0 + m;
    const float val = acc[v] + bi;
    if (!VTRANS) {
      Y[(size_t)row * DIM + col0 + nl] = (__bf16)val;
    } else {
      const int bb = row >> 11;        // row / NK
      const int j  = row & (NK - 1);
      const int f  = col0 + nl;
      const int h  = f >> 7;
      const int d  = f & (DH - 1);
      Y[((size_t)(bb * HEADS + h) * DH + d) * NK + j] = (__bf16)val;
    }
  }
}

// ---- Fused attention: per (batch, 32-row query tile), loop over heads ------
__global__ __launch_bounds__(256) void attn_kernel(
    const __bf16* __restrict__ qp, const __bf16* __restrict__ kp,
    const __bf16* __restrict__ vt, const int* __restrict__ len_x,
    float* __restrict__ out, float* __restrict__ attn2)
{
  __shared__ float  S[ROWS][SSTR];   // 32 x 2052 fp32 = 256.5 KB (CDNA5 LDS)
  __shared__ __bf16 Qs[ROWS][DH];    // per-head Q tile, async-staged (8 KB)

  const int blk  = blockIdx.x;
  const int b    = blk >> 7;          // NQ/ROWS = 128 tiles per batch
  const int q0   = (blk & 127) * ROWS;
  const int tid  = threadIdx.x;
  const int wave = tid >> 5;
  const int lane = tid & 31;
  const int half = lane >> 4;
  const int nl   = lane & 15;
  const int len  = len_x[b];
  const float scale = 0.03125f;       // DIM^-0.5 = 1/32

  for (int h = 0; h < HEADS; ++h) {
    // ---- async-stage the 32x128 bf16 Q tile for this head into LDS ----
    {
      const __bf16* qg = qp + (size_t)(b * NQ + q0) * DIM + h * DH;
#pragma unroll
      for (int c = tid; c < ROWS * 16; c += 256) {   // 512 x 16B chunks
        const int row = c >> 4;
        const int off = (c & 15) * 8;                // 8 bf16 per b128
        ASYNC_CP_B128(qg + (size_t)row * DIM + off, &Qs[row][off]);
      }
      wait_async0();
    }
    __syncthreads();

    // Q fragments for both 16-row blocks, resident in VGPRs (served from LDS)
    v16bf qf[2][4];
#pragma unroll
    for (int rb = 0; rb < 2; ++rb)
#pragma unroll
      for (int c = 0; c < 4; ++c)
        qf[rb][c] = frag_a_from_bf16(&Qs[rb * 16 + nl][0], c * 32, half);

    // ---- S = scale * Q K^T : each K fragment feeds 2 WMMAs ----
    for (int jt = wave; jt < NK / 16; jt += 8) {
      const int j0 = jt * 16;
      const __bf16* krow = kp + (size_t)(b * NK + j0 + nl) * DIM + h * DH;
      v8f a0 = {}, a1 = {};
#pragma unroll
      for (int c = 0; c < 4; ++c) {
        const v16bf bf = frag_b_from_bf16(krow, c * 32, half);
        a0 = wmma_bf16(qf[0][c], bf, a0);
        a1 = wmma_bf16(qf[1][c], bf, a1);
      }
      const int col = j0 + nl;
      const bool valid = col < len;
#pragma unroll
      for (int v = 0; v < 8; ++v) {
        S[v + half * 8][col]      = valid ? a0[v] * scale : -__builtin_inff();
        S[16 + v + half * 8][col] = valid ? a1[v] * scale : -__builtin_inff();
      }
    }
    __syncthreads();

    // ---- exact softmax: 8 lanes per row (wave32 xor shuffles 4/2/1) ----
    const int r  = tid >> 3;
    const int c0 = tid & 7;
    float mx = -__builtin_inff();
    for (int c = c0; c < NK; c += 8) mx = fmaxf(mx, S[r][c]);
#pragma unroll
    for (int off = 4; off; off >>= 1) mx = fmaxf(mx, __shfl_xor(mx, off, 32));
    float sum = 0.f;
    for (int c = c0; c < NK; c += 8) {
      const float p = __expf(S[r][c] - mx);
      S[r][c] = p;
      sum += p;
    }
#pragma unroll
    for (int off = 4; off; off >>= 1) sum += __shfl_xor(sum, off, 32);
    const float inv = 1.0f / sum;

    // normalize in LDS + accumulate attn2 = mean over heads (race-free RMW:
    // the same thread owns the same (r,c) elements for every head)
    float* a2 = attn2 + ((size_t)b * NQ + q0 + r) * NK;
    for (int c = c0; c < NK; c += 8) {
      const float p = S[r][c] * inv;
      S[r][c] = p;
      const float contrib = 0.125f * p;
      a2[c] = (h == 0) ? contrib : (a2[c] + contrib);
    }
    __syncthreads();

    // ---- O = P V : wave owns a 16-wide head_dim slice; V frag feeds 2 WMMAs
    const int d0 = wave * 16;
    const __bf16* vcol = vt + ((size_t)(b * HEADS + h) * DH + d0 + nl) * NK;
    v8f o0 = {}, o1 = {};
    for (int kk = 0; kk < NK; kk += 32) {
      const v16bf bf = frag_b_from_bf16(vcol, kk, half);
      v16bf a0f, a1f;
#pragma unroll
      for (int v = 0; v < 8; ++v) {
        const int kb = kk + ((v & 4) ? 16 : 0) + half * 8 + (v & 3) * 2;
        a0f[2 * v]     = (__bf16)S[nl][kb];        // stride 2052: conflict-free
        a0f[2 * v + 1] = (__bf16)S[nl][kb + 1];
        a1f[2 * v]     = (__bf16)S[16 + nl][kb];
        a1f[2 * v + 1] = (__bf16)S[16 + nl][kb + 1];
      }
      o0 = wmma_bf16(a0f, bf, o0);
      o1 = wmma_bf16(a1f, bf, o1);
    }
#pragma unroll
    for (int v = 0; v < 8; ++v) {
      const size_t base = (size_t)b * NQ + q0 + v + half * 8;
      out[(base)      * DIM + h * DH + d0 + nl] = o0[v];
      out[(base + 16) * DIM + h * DH + d0 + nl] = o1[v];
    }
    __syncthreads();  // S / Qs rewritten next head
  }
}

// ---- host launch -----------------------------------------------------------
extern "C" void kernel_launch(void* const* d_in, const int* in_sizes, int n_in,
                              void* d_out, int out_size, void* d_ws, size_t ws_size,
                              hipStream_t stream)
{
  (void)in_sizes; (void)n_in; (void)out_size; (void)ws_size;
  const float* queries = (const float*)d_in[0];
  const float* keys    = (const float*)d_in[1];
  const float* values  = (const float*)d_in[2];
  const float* Wq      = (const float*)d_in[3];
  const float* bq      = (const float*)d_in[4];
  const float* Wk      = (const float*)d_in[5];
  const float* bk      = (const float*)d_in[6];
  const float* Wv      = (const float*)d_in[7];
  const float* bv      = (const float*)d_in[8];
  const int*   len_x   = (const int*)d_in[9];

  char* ws = (char*)d_ws;
  __bf16* qp = (__bf16*)(ws);                                       // 16 MB
  __bf16* kp = (__bf16*)(ws + (size_t)NB * NQ * DIM * 2);           //  8 MB
  __bf16* vt = (__bf16*)(ws + (size_t)NB * (NQ + NK) * DIM * 2);    //  8 MB

  float* out   = (float*)d_out;
  float* attn2 = out + (size_t)NB * NQ * DIM;

  // Q: 8192 rows -> (8192/16)*64 tiles / 8 waves = 4096 blocks
  proj_kernel<false><<<dim3(4096), dim3(256), 0, stream>>>(queries, Wq, bq, qp);
  // K: 4096 rows -> 2048 blocks
  proj_kernel<false><<<dim3(2048), dim3(256), 0, stream>>>(keys, Wk, bk, kp);
  // V: transposed store [b][h][d][n2]
  proj_kernel<true ><<<dim3(2048), dim3(256), 0, stream>>>(values, Wv, bv, vt);
  // attention: B * NQ/ROWS = 256 blocks
  attn_kernel<<<dim3(256), dim3(256), 0, stream>>>(qp, kp, vt, len_x, out, attn2);
}
